// HiddenMarkovModel_84791244357823
// MI455X (gfx1250) — compile-verified, matrix operand order checked by reading, not run on hardware
//
#include <hip/hip_runtime.h>

typedef __attribute__((ext_vector_type(16))) _Float16 v16h;
typedef __attribute__((ext_vector_type(8)))  float    v8f;

#define HMM_B 64
#define HMM_T 2048
#define HMM_N 64
#define HMM_D 128
#define KTOT  256   // 2*D : [x^2 | x] combined K dimension

// workspace layout (float offsets)
#define WS_A      0                          // softmax(transition), linear domain, N*N
#define WS_LOGPI  (WS_A + HMM_N*HMM_N)       // log_softmax(state_priors), N
#define WS_T3     (WS_LOGPI + HMM_N)         // sum_d mu^2/var + logvar, N
#define WS_BLL    (WS_T3 + HMM_N)            // per-batch log-likelihood, B
#define WS_LOGB   (WS_BLL + HMM_B)           // emission log-probs, B*T*N

#define LOG2PI 1.8378770664093453f

__device__ __forceinline__ float neg_inf() { return -__builtin_inff(); }

// ---------------------------------------------------------------------------
// Prep: row-softmax of transition (linear domain), log_pi, t3[n].
// 1 block, 64 threads (thread i = row/state i).
// ---------------------------------------------------------------------------
__global__ void hmm_prep(const float* __restrict__ trans,
                         const float* __restrict__ priors,
                         const float* __restrict__ means,
                         const float* __restrict__ logvars,
                         float* __restrict__ ws) {
    const int i = threadIdx.x;   // 0..63
    // A[i][:] = softmax(trans[i][:])  (linear domain; entries in (0,1])
    float m = neg_inf();
    for (int j = 0; j < HMM_N; ++j) m = fmaxf(m, trans[i*HMM_N + j]);
    float s = 0.0f;
    for (int j = 0; j < HMM_N; ++j) s += __expf(trans[i*HMM_N + j] - m);
    const float inv = 1.0f / s;
    for (int j = 0; j < HMM_N; ++j)
        ws[WS_A + i*HMM_N + j] = __expf(trans[i*HMM_N + j] - m) * inv;

    // log_pi = log_softmax(priors)  (redundant per-thread scan, trivial)
    float mp = neg_inf();
    for (int j = 0; j < HMM_N; ++j) mp = fmaxf(mp, priors[j]);
    float sp = 0.0f;
    for (int j = 0; j < HMM_N; ++j) sp += __expf(priors[j] - mp);
    ws[WS_LOGPI + i] = priors[i] - mp - __logf(sp);

    // t3[i] = sum_d mu^2 * inv_var + logvar
    float t3 = 0.0f;
    for (int d = 0; d < HMM_D; ++d) {
        const float lv = logvars[i*HMM_D + d];
        const float mu = means[i*HMM_D + d];
        t3 += mu*mu*__expf(-lv) + lv;
    }
    ws[WS_T3 + i] = t3;
}

// ---------------------------------------------------------------------------
// Emission: log_b[b,t,n] = -0.5*( [x^2|x] . [inv_var|-2*mu*inv_var](n) + t3[n] + D*log2pi )
// Single f16 WMMA GEMM: M = B*T, K = 256, N = 64.
// Grid: (B*T)/64 blocks, 128 threads (4 waves). Each wave owns a 16-row M tile.
//
// The B operand is identical for every block, so we stage it in LDS in
// *fragment-major* order: Bfrag[nt][s][lane][0..15] halves. Each lane then
// fetches its whole v16h fragment as two contiguous ds_load_b128 (32B)
// instead of 16 scattered ds_load_u16 + b16 packing.
// ---------------------------------------------------------------------------
__global__ void hmm_emission(const float* __restrict__ X,
                             const float* __restrict__ means,
                             const float* __restrict__ logvars,
                             float* __restrict__ ws) {
    __shared__ _Float16 Bfrag[4 * 8 * 32 * 16];   // 32 KB, fragment-major

    const int tid = threadIdx.x;
    // cooperatively build pre-swizzled B fragments
    for (int idx = tid; idx < 4*8*32*16; idx += 128) {
        const int h    = idx & 15;
        const int ln   = (idx >> 4) & 31;
        const int s    = (idx >> 9) & 7;
        const int nt   = idx >> 12;
        const int bl16 = ln & 15;
        const int bkb  = (ln >> 4) * 8;
        const int kk   = (h < 8) ? (bkb + h) : (16 + bkb + (h - 8));
        const int g    = s*32 + kk;              // 0..255 combined-K index
        const int n    = nt*16 + bl16;           // state column
        const int d    = g & (HMM_D - 1);
        const float iv = __expf(-logvars[n*HMM_D + d]);
        const float v  = (g < HMM_D) ? iv : (-2.0f * means[n*HMM_D + d] * iv);
        Bfrag[idx] = (_Float16)v;
    }
    __syncthreads();

    const int wave  = tid >> 5;
    const int lane  = tid & 31;
    const int l16   = lane & 15;
    const int hi    = lane >> 4;        // 0 or 1
    const int kbase = hi * 8;           // K sub-offset per 16x32 f16 A layout
    const size_t rowbase = (size_t)blockIdx.x * 64 + (size_t)wave * 16;
    const float* __restrict__ xrow = X + (rowbase + (size_t)l16) * HMM_D;

    // Preload all 8 K-step A fragments for this wave's 16 rows.
    v16h afrag[8];
#pragma unroll
    for (int s = 0; s < 8; ++s) {
#pragma unroll
        for (int h = 0; h < 16; ++h) {
            const int kk = (h < 8) ? (kbase + h) : (16 + kbase + (h - 8));
            const int g  = s*32 + kk;
            const float xv = (g < HMM_D) ? xrow[g] : xrow[g - HMM_D];
            const float v  = (g < HMM_D) ? (xv * xv) : xv;
            afrag[s][h] = (_Float16)v;
        }
    }

    const float log2pi_term = (float)HMM_D * LOG2PI;

#pragma unroll
    for (int nt = 0; nt < 4; ++nt) {
        const int n = nt*16 + l16;
        v8f c = {};
#pragma unroll
        for (int s = 0; s < 8; ++s) {
            const v16h b = *(const v16h*)&Bfrag[((nt*8 + s)*32 + lane) * 16];
            c = __builtin_amdgcn_wmma_f32_16x16x32_f16(
                    /*neg_a=*/false, afrag[s], /*neg_b=*/false, b,
                    /*c_mod=*/(short)0, c, /*reuse_a=*/false, /*reuse_b=*/false);
        }
        const float add = ws[WS_T3 + n] + log2pi_term;
#pragma unroll
        for (int r = 0; r < 8; ++r) {
            const size_t mrow = rowbase + (size_t)(r + 8*hi);   // D layout: lanes 16-31 -> M = 8+r
            ws[WS_LOGB + mrow*HMM_N + n] = -0.5f * (c[r] + add);
        }
    }
}

// ---------------------------------------------------------------------------
// Scan: forward recursion over T with per-step max rescaling.
// Grid: B blocks, 64 threads (thread j = state j).
// alpha_t[j] = logb_t[j] + m + log( sum_i exp(alpha_{t-1}[i]-m) * A[i][j] )
// ---------------------------------------------------------------------------
__global__ void hmm_scan(float* __restrict__ ws) {
    __shared__ float A[HMM_N * HMM_N];   // 16 KB
    __shared__ float sh_a[HMM_N];
    __shared__ float sh_p[HMM_N];

    const int j = threadIdx.x;           // state
    const int b = blockIdx.x;            // batch

    for (int idx = j; idx < HMM_N*HMM_N; idx += HMM_N) A[idx] = ws[WS_A + idx];

    const float* __restrict__ logb = ws + WS_LOGB + (size_t)b * HMM_T * HMM_N;
    float alpha = ws[WS_LOGPI + j] + logb[j];
    __syncthreads();

    for (int t = 1; t < HMM_T; ++t) {
        sh_a[j] = alpha;
        __syncthreads();
        float m = neg_inf();
#pragma unroll 8
        for (int i = 0; i < HMM_N; ++i) m = fmaxf(m, sh_a[i]);   // broadcast reads
        sh_p[j] = __expf(alpha - m);
        __syncthreads();
        float s = 0.0f;
#pragma unroll 8
        for (int i = 0; i < HMM_N; ++i) s = fmaf(sh_p[i], A[i*HMM_N + j], s);
        alpha = logb[t*HMM_N + j] + m + __logf(s);
        // next iteration's first barrier protects sh_p readers vs sh_p rewrite
    }

    // final logsumexp over states
    sh_a[j] = alpha;
    __syncthreads();
    float m = neg_inf();
    for (int i = 0; i < HMM_N; ++i) m = fmaxf(m, sh_a[i]);
    sh_p[j] = __expf(alpha - m);
    __syncthreads();
    if (j == 0) {
        float s = 0.0f;
        for (int i = 0; i < HMM_N; ++i) s += sh_p[i];
        ws[WS_BLL + b] = m + __logf(s);
    }
}

// ---------------------------------------------------------------------------
// Deterministic final reduction over batches.
// ---------------------------------------------------------------------------
__global__ void hmm_final(const float* __restrict__ ws, float* __restrict__ out) {
    if (threadIdx.x == 0) {
        float s = 0.0f;
        for (int i = 0; i < HMM_B; ++i) s += ws[WS_BLL + i];
        out[0] = s;
    }
}

extern "C" void kernel_launch(void* const* d_in, const int* in_sizes, int n_in,
                              void* d_out, int out_size, void* d_ws, size_t ws_size,
                              hipStream_t stream) {
    const float* X       = (const float*)d_in[0];   // (B,T,D)
    const float* trans   = (const float*)d_in[1];   // (N,N)
    const float* priors  = (const float*)d_in[2];   // (N,)
    const float* means   = (const float*)d_in[3];   // (N,D)
    const float* logvars = (const float*)d_in[4];   // (N,D)
    float* out = (float*)d_out;
    float* ws  = (float*)d_ws;

    hmm_prep<<<1, HMM_N, 0, stream>>>(trans, priors, means, logvars, ws);
    hmm_emission<<<(HMM_B * HMM_T) / 64, 128, 0, stream>>>(X, means, logvars, ws);
    hmm_scan<<<HMM_B, HMM_N, 0, stream>>>(ws);
    hmm_final<<<1, 32, 0, stream>>>(ws, out);
}
